// GraphNeuralNetwork_31963146617551
// MI455X (gfx1250) — compile-verified, matrix operand order checked by reading, not run on hardware
//
#include <hip/hip_runtime.h>
#include <hip/hip_bf16.h>
#include <math.h>

typedef __attribute__((ext_vector_type(16))) _Float16 v16h;
typedef __attribute__((ext_vector_type(8)))  float    v8f;

#define NEG_SLOPE 0.2f
#define ENC_NEGINF 0x007FFFFFu   // fenc(-inf)

__device__ __forceinline__ unsigned fenc(float f) {
  unsigned u = __float_as_uint(f);
  return (u & 0x80000000u) ? ~u : (u | 0x80000000u);
}
__device__ __forceinline__ float fdec(unsigned u) {
  return __uint_as_float((u & 0x80000000u) ? (u & 0x7FFFFFFFu) : ~u);
}

// ---------------- generic fill ----------------
__global__ void k_fill_u32(unsigned* p, unsigned v, long long n) {
  long long i = (long long)blockIdx.x * blockDim.x + threadIdx.x;
  long long st = (long long)gridDim.x * blockDim.x;
  for (; i < n; i += st) p[i] = v;
}

// ---------------- mean of edge_attr over E rows ----------------
__global__ void k_mean_ea(const float* __restrict__ ea, int E, float* acc4) {
  __shared__ float red[256];
  float a0 = 0, a1 = 0, a2 = 0, a3 = 0;
  for (int e = blockIdx.x * blockDim.x + threadIdx.x; e < E;
       e += gridDim.x * blockDim.x) {
    a0 += ea[4 * e + 0]; a1 += ea[4 * e + 1];
    a2 += ea[4 * e + 2]; a3 += ea[4 * e + 3];
  }
  float v[4] = {a0, a1, a2, a3};
  for (int k = 0; k < 4; k++) {
    red[threadIdx.x] = v[k];
    __syncthreads();
    for (int s = 128; s > 0; s >>= 1) {
      if (threadIdx.x < s) red[threadIdx.x] += red[threadIdx.x + s];
      __syncthreads();
    }
    if (threadIdx.x == 0) atomicAdd(&acc4[k], red[0]);
    __syncthreads();
  }
}

// ---------------- Me[k,h] = sum_c We[k, h*64+c]*a_e[h,c]; loop alpha_e ----------------
__global__ void k_prep(const float* We0, const float* ae0,
                       const float* We1, const float* ae1,
                       const float* acc4, int E,
                       float* Me0, float* Me1, float* loop0, float* loop1) {
  int t = threadIdx.x;
  if (t < 32) {
    const float* We = (t < 16) ? We0 : We1;
    const float* ae = (t < 16) ? ae0 : ae1;
    float* Me = (t < 16) ? Me0 : Me1;
    int tt = t & 15, k = tt >> 2, h = tt & 3;
    float s = 0.f;
    for (int c = 0; c < 64; c++) s += We[k * 256 + h * 64 + c] * ae[h * 64 + c];
    Me[k * 4 + h] = s;
  }
  __syncthreads();
  if (t < 8) {
    const float* Me = (t < 4) ? Me0 : Me1;
    float* loop = (t < 4) ? loop0 : loop1;
    int h = t & 3;
    float s = 0.f;
    for (int k = 0; k < 4; k++) s += (acc4[k] / (float)E) * Me[k * 4 + h];
    loop[h] = s;
  }
}

// ---------------- layer0 node transform (K=5) + attention logits ----------------
__global__ void k_node0(const float* __restrict__ x, const float* __restrict__ W0,
                        const float* __restrict__ asrc, const float* __restrict__ adst,
                        float* __restrict__ h_lin, float* al_s, float* al_d) {
  int n = blockIdx.x, c = threadIdx.x;
  float acc = 0.f;
#pragma unroll
  for (int k = 0; k < 5; k++) acc += x[n * 5 + k] * W0[k * 256 + c];
  h_lin[(size_t)n * 256 + c] = acc;
  __shared__ float bs[256], bd[256];
  int hh = c >> 6, ci = c & 63;
  bs[c] = acc * asrc[hh * 64 + ci];
  bd[c] = acc * adst[hh * 64 + ci];
  __syncthreads();
  for (int s = 32; s > 0; s >>= 1) {
    if (ci < s) { bs[c] += bs[c + s]; bd[c] += bd[c + s]; }
    __syncthreads();
  }
  if (ci == 0) { al_s[n * 4 + hh] = bs[c]; al_d[n * 4 + hh] = bd[c]; }
}

// ---------------- layer1 attention logits (h already computed) ----------------
__global__ void k_node_al(const float* __restrict__ h_lin,
                          const float* __restrict__ asrc, const float* __restrict__ adst,
                          float* al_s, float* al_d) {
  int n = blockIdx.x, c = threadIdx.x;
  float acc = h_lin[(size_t)n * 256 + c];
  __shared__ float bs[256], bd[256];
  int hh = c >> 6, ci = c & 63;
  bs[c] = acc * asrc[hh * 64 + ci];
  bd[c] = acc * adst[hh * 64 + ci];
  __syncthreads();
  for (int s = 32; s > 0; s >>= 1) {
    if (ci < s) { bs[c] += bs[c + s]; bd[c] += bd[c + s]; }
    __syncthreads();
  }
  if (ci == 0) { al_s[n * 4 + hh] = bs[c]; al_d[n * 4 + hh] = bd[c]; }
}

// ---------------- edge logits + segment max ----------------
__global__ void k_edge_logits(const int* __restrict__ ei, const float* __restrict__ ea,
                              int E, int Etot, const float* __restrict__ Me,
                              const float* __restrict__ loopv,
                              const float* __restrict__ al_s, const float* __restrict__ al_d,
                              float* __restrict__ alpha, unsigned* __restrict__ m_enc) {
  for (int e = blockIdx.x * blockDim.x + threadIdx.x; e < Etot;
       e += gridDim.x * blockDim.x) {
    int s, d;
    float ale[4];
    if (e < E) {
      s = ei[e]; d = ei[E + e];
      float e0 = ea[4 * e + 0], e1 = ea[4 * e + 1];
      float e2 = ea[4 * e + 2], e3 = ea[4 * e + 3];
#pragma unroll
      for (int h = 0; h < 4; h++)
        ale[h] = e0 * Me[h] + e1 * Me[4 + h] + e2 * Me[8 + h] + e3 * Me[12 + h];
    } else {
      s = d = e - E;
#pragma unroll
      for (int h = 0; h < 4; h++) ale[h] = loopv[h];
    }
#pragma unroll
    for (int h = 0; h < 4; h++) {
      float a = al_s[s * 4 + h] + al_d[d * 4 + h] + ale[h];
      a = (a > 0.f) ? a : NEG_SLOPE * a;
      alpha[(size_t)e * 4 + h] = a;
      atomicMax(&m_enc[d * 4 + h], fenc(a));
    }
  }
}

// ---------------- exp(alpha - max) + segment denom ----------------
__global__ void k_edge_exp(const int* __restrict__ ei, int E, int Etot,
                           const unsigned* __restrict__ m_enc,
                           float* __restrict__ alpha, float* __restrict__ denom) {
  for (int e = blockIdx.x * blockDim.x + threadIdx.x; e < Etot;
       e += gridDim.x * blockDim.x) {
    int d = (e < E) ? ei[E + e] : e - E;
#pragma unroll
    for (int h = 0; h < 4; h++) {
      float p = __expf(alpha[(size_t)e * 4 + h] - fdec(m_enc[d * 4 + h]));
      alpha[(size_t)e * 4 + h] = p;
      atomicAdd(&denom[d * 4 + h], p);
    }
  }
}

// ---------------- weighted message scatter-add (1 block / edge) ----------------
__global__ void k_edge_agg(const int* __restrict__ ei, int E,
                           const float* __restrict__ h_lin, const float* __restrict__ p,
                           const float* __restrict__ denom, float* __restrict__ out_acc) {
  int e = blockIdx.x, c = threadIdx.x, h = c >> 6;
  int s, d;
  if (e < E) { s = ei[e]; d = ei[E + e]; }
  else       { s = d = e - E; }
  float w = p[(size_t)e * 4 + h] / (denom[d * 4 + h] + 1e-16f);
  atomicAdd(&out_acc[(size_t)d * 256 + c], h_lin[(size_t)s * 256 + c] * w);
}

// ---------------- layer0 epilogue: +bias, relu, pack f16 A-matrix layout ----------------
// A layout (16-bit, 16x32 per chunk): lane = r (kin in [0,8)|[16,24)) or 16+r,
// element j = (kin&7) + 8*(kin>=16)   [ISA 7.12.2]
__global__ void k_finalize0(const float* __restrict__ out_acc, const float* __restrict__ b,
                            _Float16* __restrict__ apack, long long total) {
  long long stride = (long long)gridDim.x * blockDim.x;
  for (long long idx = (long long)blockIdx.x * blockDim.x + threadIdx.x; idx < total;
       idx += stride) {
    int c = (int)(idx & 255);
    long long n = idx >> 8;
    float v = out_acc[idx] + b[c];
    v = (v > 0.f) ? v : 0.f;  // relu between layers
    int r = (int)(n & 15);
    long long mt = n >> 4;
    int kc = c >> 5, kin = c & 31;
    int lane = ((kin >> 3) & 1) ? (16 + r) : r;
    int j = (kin & 7) + ((kin >= 16) ? 8 : 0);
    apack[((mt * 8 + kc) * 32 + lane) * 16 + j] = (_Float16)v;
  }
}

// ---------------- pack W1 into per-lane f16 B-matrix layout ----------------
__global__ void k_w1pack(const float* __restrict__ W1, _Float16* __restrict__ bpack) {
  int idx = blockIdx.x * blockDim.x + threadIdx.x;
  if (idx >= 256 * 256) return;
  int k = idx >> 8, n = idx & 255;
  int kc = k >> 5, kin = k & 31, nt = n >> 4, nr = n & 15;
  int lane = ((kin >> 3) & 1) ? (16 + nr) : nr;
  int j = (kin & 7) + ((kin >= 16) ? 8 : 0);
  bpack[((size_t)(nt * 8 + kc) * 32 + lane) * 16 + j] = (_Float16)W1[idx];
}

// ---------------- WMMA GEMM: [N,256](f16) x [256,256](f16) -> f32 ----------------
__global__ void __launch_bounds__(256)
k_wmma_gemm(const _Float16* __restrict__ apack, const _Float16* __restrict__ bpack,
            float* __restrict__ hout, int mtiles, int N) {
  int wave = threadIdx.x >> 5, lane = threadIdx.x & 31;
  int tile = blockIdx.x * 8 + wave;
  if (tile >= mtiles * 16) return;           // whole-wave exit, EXEC stays uniform
  int nt = tile & 15, mt = tile >> 4;
  const _Float16* A = apack + ((size_t)mt * 8 * 32 + lane) * 16;
  const _Float16* B = bpack + ((size_t)nt * 8 * 32 + lane) * 16;
  v8f c = {0.f, 0.f, 0.f, 0.f, 0.f, 0.f, 0.f, 0.f};
#pragma unroll
  for (int kc = 0; kc < 8; kc++) {
    v16h a = *(const v16h*)(A + (size_t)kc * 512);
    v16h b = *(const v16h*)(B + (size_t)kc * 512);
    if (kc < 7) __builtin_prefetch(A + (size_t)(kc + 1) * 512, 0, 1);
    c = __builtin_amdgcn_wmma_f32_16x16x32_f16(false, a, false, b, (short)0, c,
                                               false, false);
  }
  int ncol = nt * 16 + (lane & 15);
  int mbase = mt * 16 + ((lane >= 16) ? 8 : 0);
#pragma unroll
  for (int v = 0; v < 8; v++) {
    int row = mbase + v;
    if (row < N) hout[(size_t)row * 256 + ncol] = c[v];
  }
}

// ---------------- layer1 epilogue: +bias in place ----------------
__global__ void k_finalize1(float* __restrict__ out_acc, const float* __restrict__ b,
                            long long total) {
  long long stride = (long long)gridDim.x * blockDim.x;
  for (long long idx = (long long)blockIdx.x * blockDim.x + threadIdx.x; idx < total;
       idx += stride)
    out_acc[idx] += b[idx & 255];
}

// ---------------- mean/max pooling, 64 nodes per block ----------------
__global__ void k_pool(const float* __restrict__ h, int N, float* mean_acc,
                       unsigned* max_enc) {
  int c = threadIdx.x;
  int base = blockIdx.x * 64;
  float sum = 0.f, mx = -INFINITY;
  for (int i = 0; i < 64; i++) {
    int n = base + i;
    if (n < N) {
      float v = h[(size_t)n * 256 + c];
      sum += v;
      mx = fmaxf(mx, v);
    }
  }
  atomicAdd(&mean_acc[c], sum);
  atomicMax(&max_enc[c], fenc(mx));
}

__global__ void k_meandiv(const float* mean_acc, int N, float* mean_pool) {
  int c = threadIdx.x;
  mean_pool[c] = mean_acc[c] / (float)N;
}

// ---------------- attention scores s[n] = h[n] . mean_pool, track max ----------------
__global__ void k_score(const float* __restrict__ h, const float* __restrict__ mean_pool,
                        float* __restrict__ s, unsigned* smax_enc) {
  __shared__ float red[256];
  int n = blockIdx.x, c = threadIdx.x;
  red[c] = h[(size_t)n * 256 + c] * mean_pool[c];
  __syncthreads();
  for (int st = 128; st > 0; st >>= 1) {
    if (c < st) red[c] += red[c + st];
    __syncthreads();
  }
  if (c == 0) {
    s[n] = red[0];
    atomicMax(smax_enc, fenc(red[0]));
  }
}

// ---------------- attention pooling accumulate ----------------
__global__ void k_att(const float* __restrict__ h, const float* __restrict__ s,
                      const unsigned* smax_enc, int N, float* att_acc, float* esum) {
  int c = threadIdx.x;
  int base = blockIdx.x * 64;
  float smax = fdec(*smax_enc);
  float acc = 0.f, el = 0.f;
  for (int i = 0; i < 64; i++) {
    int n = base + i;
    if (n < N) {
      float e_ = __expf(s[n] - smax);
      acc += e_ * h[(size_t)n * 256 + c];
      if (c == 0) el += e_;
    }
  }
  atomicAdd(&att_acc[c], acc);
  if (c == 0) atomicAdd(esum, el);
}

// ---------------- final MLP (single block of 128) ----------------
__global__ void k_mlp(const float* mean_pool, const unsigned* max_enc,
                      const float* att_acc, const float* esum,
                      const float* Wm1, const float* bm1,
                      const float* Wm2, const float* bm2,
                      const float* Wm3, const float* bm3, float* out) {
  __shared__ float comb[768], z1[128], z2[64];
  int t = threadIdx.x;
  for (int i = t; i < 768; i += 128) {
    float v;
    if (i < 256)      v = mean_pool[i];
    else if (i < 512) v = fdec(max_enc[i - 256]);
    else              v = att_acc[i - 512] / esum[0];
    comb[i] = v;
  }
  __syncthreads();
  {
    float z = bm1[t];
    for (int i = 0; i < 768; i++) z += comb[i] * Wm1[i * 128 + t];
    z1[t] = fmaxf(z, 0.f);
  }
  __syncthreads();
  if (t < 64) {
    float z = bm2[t];
    for (int i = 0; i < 128; i++) z += z1[i] * Wm2[i * 64 + t];
    z2[t] = fmaxf(z, 0.f);
  }
  __syncthreads();
  {
    float g = bm3[t];
    for (int i = 0; i < 64; i++) g += z2[i] * Wm3[i * 128 + t];
    out[t] = g;
  }
}

extern "C" void kernel_launch(void* const* d_in, const int* in_sizes, int n_in,
                              void* d_out, int out_size, void* d_ws, size_t ws_size,
                              hipStream_t stream) {
  const float* x   = (const float*)d_in[0];
  const int*   ei  = (const int*)d_in[1];
  const float* ea  = (const float*)d_in[2];
  const float* W0  = (const float*)d_in[3];
  const float* as0 = (const float*)d_in[4];
  const float* ad0 = (const float*)d_in[5];
  const float* We0 = (const float*)d_in[6];
  const float* ae0 = (const float*)d_in[7];
  const float* b0  = (const float*)d_in[8];
  const float* W1  = (const float*)d_in[9];
  const float* as1 = (const float*)d_in[10];
  const float* ad1 = (const float*)d_in[11];
  const float* We1 = (const float*)d_in[12];
  const float* ae1 = (const float*)d_in[13];
  const float* b1  = (const float*)d_in[14];
  const float* Wm1 = (const float*)d_in[15];
  const float* bm1 = (const float*)d_in[16];
  const float* Wm2 = (const float*)d_in[17];
  const float* bm2 = (const float*)d_in[18];
  const float* Wm3 = (const float*)d_in[19];
  const float* bm3 = (const float*)d_in[20];
  float* out = (float*)d_out;

  const int N = in_sizes[0] / 5;
  const int E = in_sizes[1] / 2;
  const int Etot = E + N;
  const int mtiles = (N + 15) / 16;

  char* ws = (char*)d_ws;
  size_t off = 0;
  auto carve = [&](size_t bytes) -> char* {
    char* p = ws + off;
    off = (off + bytes + 255) & ~(size_t)255;
    return p;
  };
  float*    h_lin    = (float*)carve((size_t)N * 256 * 4);
  float*    out_acc  = (float*)carve((size_t)N * 256 * 4);
  _Float16* apack    = (_Float16*)carve((size_t)mtiles * 16 * 256 * 2);
  _Float16* bpack    = (_Float16*)carve((size_t)256 * 256 * 2);
  float*    alpha    = (float*)carve((size_t)Etot * 4 * 4);
  float*    al_s     = (float*)carve((size_t)N * 4 * 4);
  float*    al_d     = (float*)carve((size_t)N * 4 * 4);
  unsigned* m_enc    = (unsigned*)carve((size_t)N * 4 * 4);
  float*    denom    = (float*)carve((size_t)N * 4 * 4);
  float*    sbuf     = (float*)carve((size_t)N * 4);
  float*    acc4     = (float*)carve(4 * 4);
  float*    Me0      = (float*)carve(16 * 4);
  float*    Me1      = (float*)carve(16 * 4);
  float*    loop0    = (float*)carve(4 * 4);
  float*    loop1    = (float*)carve(4 * 4);
  float*    mean_acc = (float*)carve(256 * 4);
  float*    mean_pool= (float*)carve(256 * 4);
  unsigned* max_enc  = (unsigned*)carve(256 * 4);
  float*    att_acc  = (float*)carve(256 * 4);
  float*    esum     = (float*)carve(4);
  unsigned* smax     = (unsigned*)carve(4);
  (void)ws_size; (void)n_in; (void)out_size;

  // ---- global small-accumulator init ----
  k_fill_u32<<<1, 256, 0, stream>>>((unsigned*)acc4, 0u, 4);
  k_fill_u32<<<1, 256, 0, stream>>>((unsigned*)mean_acc, 0u, 256);
  k_fill_u32<<<1, 256, 0, stream>>>((unsigned*)att_acc, 0u, 256);
  k_fill_u32<<<1, 256, 0, stream>>>(max_enc, ENC_NEGINF, 256);
  k_fill_u32<<<1, 64, 0, stream>>>((unsigned*)esum, 0u, 1);
  k_fill_u32<<<1, 64, 0, stream>>>(smax, ENC_NEGINF, 1);

  // ---- edge-attr mean + folded attention matrices ----
  k_mean_ea<<<256, 256, 0, stream>>>(ea, E, acc4);
  k_prep<<<1, 64, 0, stream>>>(We0, ae0, We1, ae1, acc4, E, Me0, Me1, loop0, loop1);

  // ================= layer 0 =================
  k_node0<<<N, 256, 0, stream>>>(x, W0, as0, ad0, h_lin, al_s, al_d);
  k_fill_u32<<<256, 256, 0, stream>>>(m_enc, ENC_NEGINF, (long long)N * 4);
  k_fill_u32<<<256, 256, 0, stream>>>((unsigned*)denom, 0u, (long long)N * 4);
  k_fill_u32<<<1024, 256, 0, stream>>>((unsigned*)out_acc, 0u, (long long)N * 256);
  k_edge_logits<<<4096, 256, 0, stream>>>(ei, ea, E, Etot, Me0, loop0, al_s, al_d,
                                          alpha, m_enc);
  k_edge_exp<<<4096, 256, 0, stream>>>(ei, E, Etot, m_enc, alpha, denom);
  k_edge_agg<<<Etot, 256, 0, stream>>>(ei, E, h_lin, alpha, denom, out_acc);
  k_finalize0<<<2048, 256, 0, stream>>>(out_acc, b0, apack, (long long)N * 256);

  // ================= layer 1 dense transform via WMMA =================
  k_w1pack<<<256, 256, 0, stream>>>(W1, bpack);
  {
    int total_tiles = mtiles * 16;
    int blocks = (total_tiles + 7) / 8;
    k_wmma_gemm<<<blocks, 256, 0, stream>>>(apack, bpack, h_lin, mtiles, N);
  }
  k_node_al<<<N, 256, 0, stream>>>(h_lin, as1, ad1, al_s, al_d);

  // ================= layer 1 edge phase =================
  k_fill_u32<<<256, 256, 0, stream>>>(m_enc, ENC_NEGINF, (long long)N * 4);
  k_fill_u32<<<256, 256, 0, stream>>>((unsigned*)denom, 0u, (long long)N * 4);
  k_fill_u32<<<1024, 256, 0, stream>>>((unsigned*)out_acc, 0u, (long long)N * 256);
  k_edge_logits<<<4096, 256, 0, stream>>>(ei, ea, E, Etot, Me1, loop1, al_s, al_d,
                                          alpha, m_enc);
  k_edge_exp<<<4096, 256, 0, stream>>>(ei, E, Etot, m_enc, alpha, denom);
  k_edge_agg<<<Etot, 256, 0, stream>>>(ei, E, h_lin, alpha, denom, out_acc);
  k_finalize1<<<2048, 256, 0, stream>>>(out_acc, b1, (long long)N * 256);

  // ================= pooling + MLP =================
  k_pool<<<(N + 63) / 64, 256, 0, stream>>>(out_acc, N, mean_acc, max_enc);
  k_meandiv<<<1, 256, 0, stream>>>(mean_acc, N, mean_pool);
  k_score<<<N, 256, 0, stream>>>(out_acc, mean_pool, sbuf, smax);
  k_att<<<(N + 63) / 64, 256, 0, stream>>>(out_acc, sbuf, smax, N, att_acc, esum);
  k_mlp<<<1, 128, 0, stream>>>(mean_pool, max_enc, att_acc, esum, Wm1, bm1, Wm2, bm2,
                               Wm3, bm3, out);
}
